// GTEncoderLayer_47614007443925
// MI455X (gfx1250) — matrix operand order repaired and checked by reading, not compiled, not run on hardware
//
#include <hip/hip_runtime.h>
#include <cstdint>
#include <cstddef>

// ---------------------------------------------------------------------------
// GT encoder layer for MI455X (gfx1250).  B=2, N=512, D=64, H=8, Dh=128.
// Edge tensor dominates (128MB).  Edge MLP (17 GFLOP) + E/G projection run on
// v_wmma_f32_16x16x32_f16 with B-operands pre-packed in LDS in fragment order
// (ds_load_b128 instead of ds_load_u16 gathers).  fp32 LN/softmax in VALU.
// ---------------------------------------------------------------------------

#define BB 2
#define NN 512
#define DD 64
#define HH 8
#define DH 128
#define EPS 1e-5f

typedef _Float16 f16;
typedef __attribute__((ext_vector_type(16))) _Float16 v16h;
typedef __attribute__((ext_vector_type(8)))  _Float16 v8h;
typedef __attribute__((ext_vector_type(8)))  float    v8f;

__device__ __forceinline__ v8f wmma16(v16h a, v16h b, v8f c) {
  return __builtin_amdgcn_wmma_f32_16x16x32_f16(
      /*neg_a=*/false, a, /*neg_b=*/false, b,
      /*c_mod=*/(short)0, c, /*reuse_a=*/false, /*reuse_b=*/false);
}

// Wave-local LDS producer->consumer sync (staging tiles are wave-private, so a
// full s_barrier is unnecessary; CDNA5 split counter wait is enough).
__device__ __forceinline__ void wave_lds_sync() {
  __builtin_amdgcn_wave_barrier();
  asm volatile("s_wait_dscnt 0x0" ::: "memory");
  __builtin_amdgcn_wave_barrier();
}

__device__ __forceinline__ float fast_sigmoid(float x) {
  return __builtin_amdgcn_rcpf(1.f + __expf(-x));
}

// A fragment (16-bit A 16x32), contiguous k = 16h + s.  Strides are multiples of 8 f16
// so both halves are 16B-aligned -> two ds_load_b128 per fragment.
__device__ __forceinline__ v16h frag_a_f16(const f16* src, int stride, int k0, int lane) {
  int m  = lane & 15;
  int kb = (lane < 16) ? 0 : 16;
  const f16* p = src + m * stride + k0 + kb;
  v8h lo = *(const v8h*)(p);
  v8h hi = *(const v8h*)(p + 8);
  return __builtin_shufflevector(lo, hi, 0,1,2,3,4,5,6,7,8,9,10,11,12,13,14,15);
}

// Packed B fragment: layout [half(2)][lane(32)][8 f16] = 512 f16 per fragment.
// Lane stride is 16B -> wave-wide ds_load_b128 at the 2-cycle minimum.
__device__ __forceinline__ v16h load_fragB(const f16* fragbase, int lane) {
  const v8h* p = (const v8h*)fragbase;
  v8h lo = p[lane];
  v8h hi = p[32 + lane];
  return __builtin_shufflevector(lo, hi, 0,1,2,3,4,5,6,7,8,9,10,11,12,13,14,15);
}

// ---------------------------------------------------------------------------
// K1: node prep.  n1 = LN(n); Q/K/V = n1 @ Wq/Wk/Wv.  grid=B*N, block=64.
// ---------------------------------------------------------------------------
__global__ void gt_node_prep(const float* __restrict__ n,
                             const float* __restrict__ Wq, const float* __restrict__ Wk,
                             const float* __restrict__ Wv,
                             const float* __restrict__ g, const float* __restrict__ bbias,
                             float* __restrict__ n1_ws, float* __restrict__ q_ws,
                             float* __restrict__ k_ws, float* __restrict__ v_ws) {
  __shared__ float row[DD];
  __shared__ float n1r[DD];
  int bi = blockIdx.x;
  int d  = threadIdx.x;
  row[d] = n[(size_t)bi * DD + d];
  __syncthreads();
  float s = 0.f, ss = 0.f;
#pragma unroll 8
  for (int k = 0; k < DD; ++k) { float v = row[k]; s += v; ss += v * v; }
  float mean = s * (1.f / DD);
  float inv  = rsqrtf(ss * (1.f / DD) - mean * mean + EPS);
  float v1   = (row[d] - mean) * inv * g[d] + bbias[d];
  n1r[d] = v1;
  n1_ws[(size_t)bi * DD + d] = v1;
  __syncthreads();
  float q = 0.f, kk = 0.f, vv = 0.f;
#pragma unroll 8
  for (int k = 0; k < DD; ++k) {
    float a = n1r[k];
    q  += a * Wq[k * DD + d];
    kk += a * Wk[k * DD + d];
    vv += a * Wv[k * DD + d];
  }
  q_ws[(size_t)bi * DD + d] = q;
  k_ws[(size_t)bi * DD + d] = kk;
  v_ws[(size_t)bi * DD + d] = vv;
}

// ---------------------------------------------------------------------------
// K2: edge pass A.  Per (b,i): LN edge rows, WMMA against packed [We|Wg]
// (64x16), write E[b,i,j,h] (ws), reduce sigmoid(G) over j -> cent[b,h,i].
// grid = B*N, block = 256 (8 waves, 4 tiles of 16 rows each).
// ---------------------------------------------------------------------------
#define K2_TILE_STRIDE 72   // f16, multiple of 8 -> 16B-aligned rows
__global__ void gt_edge_eg(const float* __restrict__ e,
                           const float* __restrict__ We, const float* __restrict__ Wg,
                           const float* __restrict__ g1, const float* __restrict__ bb1,
                           float* __restrict__ E_ws, float* __restrict__ cent_ws) {
  extern __shared__ char smem[];
  f16*   wcombf = (f16*)smem;                                 // 2 frags x 512 f16
  f16*   tiles  = (f16*)(smem + 2 * 512 * 2);                 // 8 waves x 16 x 72
  float* sg1    = (float*)(smem + 2 * 512 * 2 + 8 * 16 * K2_TILE_STRIDE * 2);
  float* sb1    = sg1 + DD;
  float* sGsum  = sb1 + DD;                                   // 8

  int tid = threadIdx.x, lane = tid & 31, wave = tid >> 5;
  int bi = blockIdx.x, b = bi >> 9, i = bi & (NN - 1);

  // pack [We|Wg] into WMMA-B fragment order (frag = k-step kk)
  for (int x = tid; x < 2 * 512; x += 256) {
    int f = x >> 9;
    int r = x & 511;
    int half = r >> 8;
    int l = (r >> 3) & 31;
    int t = (r & 7) + half * 8;
    int nn = l & 15;
    int k  = f * 32 + ((l < 16) ? 0 : 16) + t;
    float v = (nn < 8) ? We[k * HH + nn] : Wg[k * HH + (nn - 8)];
    wcombf[x] = (f16)v;
  }
  if (tid < DD) { sg1[tid] = g1[tid]; sb1[tid] = bb1[tid]; }
  if (tid < HH) sGsum[tid] = 0.f;
  __syncthreads();

  f16* myt  = tiles + wave * (16 * K2_TILE_STRIDE);
  int  row  = lane >> 1, half = lane & 1;
  int  nsel = lane & 15;
  int  mbase = (lane < 16) ? 0 : 8;
  float gacc = 0.f;

  for (int t = 0; t < 4; ++t) {
    int j0 = (wave * 4 + t) * 16;
    const float* src = e + (((size_t)(b * NN + i)) * NN + (j0 + row)) * DD + half * 32;
    float x[32];
    float s = 0.f, ss = 0.f;
#pragma unroll
    for (int c = 0; c < 32; ++c) { float v = src[c]; x[c] = v; s += v; ss += v * v; }
    s  += __shfl_xor(s, 1, 32);
    ss += __shfl_xor(ss, 1, 32);
    float mean = s * (1.f / DD);
    float inv  = rsqrtf(ss * (1.f / DD) - mean * mean + EPS);
    f16* dst = myt + row * K2_TILE_STRIDE + half * 32;
#pragma unroll
    for (int c8 = 0; c8 < 4; ++c8) {
      v8h pk;
#pragma unroll
      for (int jj = 0; jj < 8; ++jj) {
        int c = c8 * 8 + jj;
        int d = half * 32 + c;
        pk[jj] = (f16)((x[c] - mean) * inv * sg1[d] + sb1[d]);
      }
      ((v8h*)dst)[c8] = pk;
    }
    wave_lds_sync();

    v8f acc = {};
    acc = wmma16(frag_a_f16(myt, K2_TILE_STRIDE, 0, lane),  load_fragB(wcombf,       lane), acc);
    acc = wmma16(frag_a_f16(myt, K2_TILE_STRIDE, 32, lane), load_fragB(wcombf + 512, lane), acc);

    if (nsel < HH) {  // edge-bias E columns
#pragma unroll
      for (int r = 0; r < 8; ++r) {
        int j = j0 + mbase + r;
        E_ws[(((size_t)(b * NN + i)) * NN + j) * HH + nsel] = acc[r];
      }
    } else {          // gate columns: sum sigmoid over j
#pragma unroll
      for (int r = 0; r < 8; ++r) gacc += fast_sigmoid(acc[r]);
    }
    wave_lds_sync();
  }
  if (nsel >= HH) atomicAdd(&sGsum[nsel - HH], gacc);
  __syncthreads();
  if (tid < HH) cent_ws[(b * HH + tid) * NN + i] = log1pf(sGsum[tid]);
}

// ---------------------------------------------------------------------------
// K3: attention.  grid = B*H, block = 512 (one row i per thread).
// Online softmax; writes Elog[b,i,j,h] and Vc[b,h,i,:] (already * cent).
// ---------------------------------------------------------------------------
__global__ void gt_attn(const float* __restrict__ q_ws, const float* __restrict__ k_ws,
                        const float* __restrict__ v_ws, const float* __restrict__ E_ws,
                        const float* __restrict__ cent_ws,
                        float* __restrict__ Elog_ws, float* __restrict__ vc_ws) {
  __shared__ float Kh[NN * 8];
  __shared__ float Vh[NN * 8];
  int b = blockIdx.x >> 3, h = blockIdx.x & 7;
  int i = threadIdx.x;
  for (int x = i; x < NN * 8; x += NN) {
    int j = x >> 3, t = x & 7;
    Kh[x] = k_ws[((size_t)(b * NN + j)) * DD + h * 8 + t];
    Vh[x] = v_ws[((size_t)(b * NN + j)) * DD + h * 8 + t];
  }
  __syncthreads();
  float Qr[8];
#pragma unroll
  for (int t = 0; t < 8; ++t) Qr[t] = q_ws[((size_t)(b * NN + i)) * DD + h * 8 + t];

  float m0 = -1e30f, s0 = 0.f, acc[8] = {0, 0, 0, 0, 0, 0, 0, 0};
  const float* Erow = E_ws + ((size_t)(b * NN + i)) * NN * HH + h;
  float*       Lrow = Elog_ws + ((size_t)(b * NN + i)) * NN * HH + h;
  for (int j = 0; j < NN; ++j) {
    float a = 0.f;
#pragma unroll
    for (int t = 0; t < 8; ++t) a += Qr[t] * Kh[j * 8 + t];
    a *= 0.35355339059327373f;                 // 1/sqrt(dk), dk = 8
    a = fminf(fmaxf(a, -5.f), 5.f);
    float l = a + Erow[(size_t)j * HH];
    Lrow[(size_t)j * HH] = l;
    float p;
    if (l > m0) {
      float c = __expf(m0 - l);
      s0 *= c;
#pragma unroll
      for (int t = 0; t < 8; ++t) acc[t] *= c;
      m0 = l;
      p = 1.f;
    } else {
      p = __expf(l - m0);
    }
    s0 += p;
#pragma unroll
    for (int t = 0; t < 8; ++t) acc[t] += p * Vh[j * 8 + t];
  }
  float scale = cent_ws[(b * HH + h) * NN + i] * __builtin_amdgcn_rcpf(s0);
#pragma unroll
  for (int t = 0; t < 8; ++t)
    vc_ws[(((size_t)(b * HH + h)) * NN + i) * 8 + t] = acc[t] * scale;
}

// ---------------------------------------------------------------------------
// K4: node finish.  n_attn = Vc@Wo_n; n2 = n1+n_attn; LN; MLP; residual.
// grid = B*N, block = 64.
// ---------------------------------------------------------------------------
__global__ void gt_node_finish(const float* __restrict__ vc_ws, const float* __restrict__ n1_ws,
                               const float* __restrict__ Won,
                               const float* __restrict__ g2, const float* __restrict__ bb2,
                               const float* __restrict__ W1, const float* __restrict__ b1,
                               const float* __restrict__ W2, const float* __restrict__ b2,
                               float* __restrict__ nout) {
  __shared__ float vrow[DD];
  __shared__ float n2r[DD];
  __shared__ float n3r[DD];
  __shared__ float hrow[DH];
  int bi = blockIdx.x, b = bi >> 9, i = bi & (NN - 1);
  int d = threadIdx.x, h = d >> 3, t = d & 7;
  vrow[d] = vc_ws[(((size_t)(b * HH + h)) * NN + i) * 8 + t];
  __syncthreads();
  float na = 0.f;
#pragma unroll 8
  for (int k = 0; k < DD; ++k) na += vrow[k] * Won[k * DD + d];
  float n2 = n1_ws[(size_t)bi * DD + d] + na;
  n2r[d] = n2;
  __syncthreads();
  float s = 0.f, ss = 0.f;
#pragma unroll 8
  for (int k = 0; k < DD; ++k) { float v = n2r[k]; s += v; ss += v * v; }
  float mean = s * (1.f / DD);
  float inv  = rsqrtf(ss * (1.f / DD) - mean * mean + EPS);
  float n3   = (n2 - mean) * inv * g2[d] + bb2[d];
  n3r[d] = n3;
  __syncthreads();
#pragma unroll
  for (int rep = 0; rep < 2; ++rep) {
    int jj = d + rep * DD;
    float hv = b1[jj];
#pragma unroll 8
    for (int k = 0; k < DD; ++k) hv += n3r[k] * W1[k * DH + jj];
    hrow[jj] = fmaxf(hv, 0.f);
  }
  __syncthreads();
  float o = b2[d];
#pragma unroll 8
  for (int k = 0; k < DH; ++k) o += hrow[k] * W2[k * DD + d];
  nout[(size_t)bi * DD + d] = n3 + fmaxf(o, 0.f);
}

// ---------------------------------------------------------------------------
// K5: edge pass B (heavy).  Per (b,i): re-LN e rows, e_attn from Elog@Wo_e,
// second LN, then 64->128->64 MLP entirely on WMMA, residual, store.
// grid = B*N, block = 256 (8 waves x 4 tiles of 16 edge rows).
// ---------------------------------------------------------------------------
#define K5_E3_STRIDE 72    // f16, 144B rows -> 16B aligned
#define K5_H_STRIDE  136   // f16, 272B rows -> 16B aligned
__global__ void gt_edge_mlp(const float* __restrict__ e, const float* __restrict__ Elog_ws,
                            const float* __restrict__ Woe,
                            const float* __restrict__ g1, const float* __restrict__ bb1,
                            const float* __restrict__ g2, const float* __restrict__ bb2,
                            const float* __restrict__ W1, const float* __restrict__ b1,
                            const float* __restrict__ W2, const float* __restrict__ b2,
                            float* __restrict__ eout) {
  extern __shared__ char smem[];
  f16*   w1f  = (f16*)smem;                               // 16 frags x 512 f16
  f16*   w2f  = w1f + 16 * 512;                           // 16 frags x 512 f16
  float* swoe = (float*)(w2f + 16 * 512);                 // 8 x 64
  float* sb1  = swoe + HH * DD;                           // 128
  float* sb2  = sb1 + DH;                                 // 64
  float* sg1  = sb2 + DD;
  float* sbb1 = sg1 + DD;
  float* sg2  = sbb1 + DD;
  float* sbb2 = sg2 + DD;
  f16*   pwave = (f16*)(sbb2 + DD);                       // per-wave staging

  int tid = threadIdx.x, lane = tid & 31, wave = tid >> 5;
  int bi = blockIdx.x, b = bi >> 9, i = bi & (NN - 1);

  // pack W1 (64x128) into fragment order: frag f = nt*2 + kk
  for (int x = tid; x < 16 * 512; x += 256) {
    int f = x >> 9, r = x & 511;
    int half = r >> 8, l = (r >> 3) & 31;
    int t = (r & 7) + half * 8;
    int nt = f >> 1, kk = f & 1;
    int nn = nt * 16 + (l & 15);
    int k  = kk * 32 + ((l < 16) ? 0 : 16) + t;
    w1f[x] = (f16)W1[k * DH + nn];
  }
  // pack W2 (128x64) into fragment order: frag f = nt*4 + kk
  for (int x = tid; x < 16 * 512; x += 256) {
    int f = x >> 9, r = x & 511;
    int half = r >> 8, l = (r >> 3) & 31;
    int t = (r & 7) + half * 8;
    int nt = f >> 2, kk = f & 3;
    int nn = nt * 16 + (l & 15);
    int k  = kk * 32 + ((l < 16) ? 0 : 16) + t;
    w2f[x] = (f16)W2[k * DD + nn];
  }
  for (int x = tid; x < HH * DD; x += 256) swoe[x] = Woe[x];
  if (tid < DH) sb1[tid] = b1[tid];
  if (tid < DD) {
    sb2[tid] = b2[tid];
    sg1[tid] = g1[tid]; sbb1[tid] = bb1[tid];
    sg2[tid] = g2[tid]; sbb2[tid] = bb2[tid];
  }
  __syncthreads();

  f16* e3h = pwave + wave * (16 * K5_E3_STRIDE + 16 * K5_H_STRIDE);
  f16* hh  = e3h + 16 * K5_E3_STRIDE;

  int row = lane >> 1, half = lane & 1;
  int nsel = lane & 15;
  int mbase = (lane < 16) ? 0 : 8;
  size_t rowbase = ((size_t)(b * NN + i)) * NN;

  for (int t = 0; t < 4; ++t) {
    int j0 = (wave * 4 + t) * 16;
    size_t erow = rowbase + (j0 + row);
    const float* src = e + erow * DD + half * 32;
    if (t < 3)
      __builtin_prefetch(e + (erow + 16) * DD + half * 32, 0, 0);

    // ---- LN1 -> e1 ----
    float x[32];
    float s = 0.f, ss = 0.f;
#pragma unroll
    for (int c = 0; c < 32; ++c) { float v = src[c]; x[c] = v; s += v; ss += v * v; }
    s  += __shfl_xor(s, 1, 32);
    ss += __shfl_xor(ss, 1, 32);
    float mean = s * (1.f / DD);
    float inv  = rsqrtf(ss * (1.f / DD) - mean * mean + EPS);
#pragma unroll
    for (int c = 0; c < 32; ++c) {
      int d = half * 32 + c;
      x[c] = (x[c] - mean) * inv * sg1[d] + sbb1[d];
    }

    // ---- e_attn = sum_h Elog[b,i,j,h] * Wo_e[h,:] ; e2 = e1 + e_attn ----
    const float* el = Elog_ws + erow * HH;
    float eh[HH];
#pragma unroll
    for (int h = 0; h < HH; ++h) eh[h] = el[h];
#pragma unroll
    for (int c = 0; c < 32; ++c) {
      int d = half * 32 + c;
      float ea = 0.f;
#pragma unroll
      for (int h = 0; h < HH; ++h) ea += eh[h] * swoe[h * DD + d];
      x[c] += ea;
    }

    // ---- LN2 -> e3 (stage to LDS as f16, vectorized b128 stores) ----
    s = 0.f; ss = 0.f;
#pragma unroll
    for (int c = 0; c < 32; ++c) { s += x[c]; ss += x[c] * x[c]; }
    s  += __shfl_xor(s, 1, 32);
    ss += __shfl_xor(ss, 1, 32);
    mean = s * (1.f / DD);
    inv  = rsqrtf(ss * (1.f / DD) - mean * mean + EPS);
    f16* dst = e3h + row * K5_E3_STRIDE + half * 32;
#pragma unroll
    for (int c8 = 0; c8 < 4; ++c8) {
      v8h pk;
#pragma unroll
      for (int jj = 0; jj < 8; ++jj) {
        int c = c8 * 8 + jj;
        int d = half * 32 + c;
        pk[jj] = (f16)((x[c] - mean) * inv * sg2[d] + sbb2[d]);
      }
      ((v8h*)dst)[c8] = pk;
    }
    wave_lds_sync();

    // ---- layer 1: h = relu(e3 @ W1 + b1)   [16 x 128] ----
    v16h a0 = frag_a_f16(e3h, K5_E3_STRIDE, 0, lane);
    v16h a1 = frag_a_f16(e3h, K5_E3_STRIDE, 32, lane);
#pragma unroll
    for (int nt = 0; nt < 8; ++nt) {
      v8f acc = {};
      acc = wmma16(a0, load_fragB(w1f + (nt * 2 + 0) * 512, lane), acc);
      acc = wmma16(a1, load_fragB(w1f + (nt * 2 + 1) * 512, lane), acc);
      int nn = nt * 16 + nsel;
      float bv = sb1[nn];
#pragma unroll
      for (int r = 0; r < 8; ++r) {
        float v = acc[r] + bv;
        hh[(mbase + r) * K5_H_STRIDE + nn] = (f16)fmaxf(v, 0.f);
      }
    }
    wave_lds_sync();

    // ---- layer 2: out = e3 + relu(h @ W2 + b2)   [16 x 64] ----
    v16h ha0 = frag_a_f16(hh, K5_H_STRIDE, 0, lane);
    v16h ha1 = frag_a_f16(hh, K5_H_STRIDE, 32, lane);
    v16h ha2 = frag_a_f16(hh, K5_H_STRIDE, 64, lane);
    v16h ha3 = frag_a_f16(hh, K5_H_STRIDE, 96, lane);
    float* outbase = eout + (rowbase + j0) * DD;
#pragma unroll
    for (int nt = 0; nt < 4; ++nt) {
      v8f acc = {};
      acc = wmma16(ha0, load_fragB(w2f + (nt * 4 + 0) * 512, lane), acc);
      acc = wmma16(ha1, load_fragB(w2f + (nt * 4 + 1) * 512, lane), acc);
      acc = wmma16(ha2, load_fragB(w2f + (nt * 4 + 2) * 512, lane), acc);
      acc = wmma16(ha3, load_fragB(w2f + (nt * 4 + 3) * 512, lane), acc);
      int nn = nt * 16 + nsel;
      float bv = sb2[nn];
#pragma unroll
      for (int r = 0; r < 8; ++r) {
        int m = mbase + r;
        float v = fmaxf(acc[r] + bv, 0.f);
        outbase[(size_t)m * DD + nn] = (float)e3h[m * K5_E3_STRIDE + nn] + v;
      }
    }
    wave_lds_sync();
  }
}

// ---------------------------------------------------------------------------
// Launch
// ---------------------------------------------------------------------------
extern "C" void kernel_launch(void* const* d_in, const int* in_sizes, int n_in,
                              void* d_out, int out_size, void* d_ws, size_t ws_size,
                              hipStream_t stream) {
  (void)in_sizes; (void)n_in; (void)out_size; (void)ws_size;
  const float* n_in_p   = (const float*)d_in[0];
  const float* e_in_p   = (const float*)d_in[1];
  const float* Wq       = (const float*)d_in[2];
  const float* Wk       = (const float*)d_in[3];
  const float* Wv       = (const float*)d_in[4];
  const float* Wo_n     = (const float*)d_in[5];
  const float* We       = (const float*)d_in[6];
  const float* Wg       = (const float*)d_in[7];
  const float* Wo_e     = (const float*)d_in[8];
  const float* ln_n1_g  = (const float*)d_in[9];
  const float* ln_n1_b  = (const float*)d_in[10];
  const float* ln_e1_g  = (const float*)d_in[11];
  const float* ln_e1_b  = (const float*)d_in[12];
  const float* ln_n2_g  = (const float*)d_in[13];
  const float* ln_n2_b  = (const float*)d_in[14];
  const float* ln_e2_g  = (const float*)d_in[15];
  const float* ln_e2_b  = (const float*)d_in[16];
  const float* mlp_n_w1 = (const float*)d_in[17];
  const float* mlp_n_b1 = (const float*)d_in[18];
  const float* mlp_n_w2 = (const float*)d_in[19];
  const float* mlp_n_b2 = (const float*)d_in[20];
  const float* mlp_e_w1 = (const float*)d_in[21];
  const float* mlp_e_b1 = (const float*)d_in[22];
  const float* mlp_e_w2 = (const float*)d_in[23];
  const float* mlp_e_b2 = (const float*)d_in[24];

  float* nout = (float*)d_out;                       // [B,N,D]
  float* eout = nout + (size_t)BB * NN * DD;         // [B,N,N,D]

  float* w       = (float*)d_ws;
  float* n1_ws   = w;                                     // 65536
  float* q_ws    = n1_ws + (size_t)BB * NN * DD;          // 65536
  float* k_ws    = q_ws + (size_t)BB * NN * DD;
  float* v_ws    = k_ws + (size_t)BB * NN * DD;
  float* vc_ws   = v_ws + (size_t)BB * NN * DD;           // B*H*N*dk = 65536
  float* cent_ws = vc_ws + (size_t)BB * HH * NN * 8;      // 8192
  float* E_ws    = cent_ws + (size_t)BB * HH * NN;        // B*N*N*H = 4194304
  float* Elog_ws = E_ws + (size_t)BB * NN * NN * HH;      // 4194304

  size_t smem_k2 = (size_t)2 * 512 * 2                       // packed [We|Wg] frags
                 + (size_t)8 * 16 * K2_TILE_STRIDE * 2       // per-wave e1 tiles
                 + (size_t)(2 * DD + HH) * 4;                // ln params + Gsum
  size_t smem_k5 = (size_t)16 * 512 * 2 + (size_t)16 * 512 * 2   // packed W1/W2 frags
                 + (size_t)(HH * DD + DH + DD + 4 * DD) * 4      // Woe, biases, ln params
                 + (size_t)8 * (16 * K5_E3_STRIDE + 16 * K5_H_STRIDE) * 2;

  gt_node_prep<<<BB * NN, DD, 0, stream>>>(n_in_p, Wq, Wk, Wv, ln_n1_g, ln_n1_b,
                                           n1_ws, q_ws, k_ws, v_ws);
  gt_edge_eg<<<BB * NN, 256, smem_k2, stream>>>(e_in_p, We, Wg, ln_e1_g, ln_e1_b,
                                                E_ws, cent_ws);
  gt_attn<<<BB * HH, NN, 0, stream>>>(q_ws, k_ws, v_ws, E_ws, cent_ws, Elog_ws, vc_ws);
  gt_node_finish<<<BB * NN, DD, 0, stream>>>(vc_ws, n1_ws, Wo_n, ln_n2_g, ln_n2_b,
                                             mlp_n_w1, mlp_n_b1, mlp_n_w2, mlp_n_b2, nout);
  gt_edge_mlp<<<BB * NN, 256, smem_k5, stream>>>(e_in_p, Elog_ws, Wo_e,
                                                 ln_e1_g, ln_e1_b, ln_e2_g, ln_e2_b,
                                                 mlp_e_w1, mlp_e_b1, mlp_e_w2, mlp_e_b2, eout);
}
